// RGCN_37838661878058
// MI455X (gfx1250) — matrix operand order repaired and checked by reading, not compiled, not run on hardware
//
#include <hip/hip_runtime.h>
#include <hip/hip_bf16.h>

#define N_NODES 50000
#define N_EDGES 800000
#define BATCH   16384

typedef __bf16 bf16;
typedef __attribute__((ext_vector_type(16))) __bf16 v16bf;
typedef __attribute__((ext_vector_type(8)))  __bf16 v8bf;
typedef __attribute__((ext_vector_type(8)))  float  v8f;
typedef __attribute__((ext_vector_type(4)))  float  v4f;
typedef __attribute__((ext_vector_type(4)))  unsigned int v4u;
typedef __attribute__((ext_vector_type(8)))  int    v8i;
typedef __attribute__((ext_vector_type(4)))  int    v4i;

union V16 { v16bf v; bf16 e[16]; v8bf h[2]; };

// native f32 -> bf16 (RNE); backend packs pairs via v_cvt_pk_bf16_f32
__device__ __forceinline__ bf16 to_bf16(float f) { return static_cast<bf16>(f); }

// convert 8 consecutive f32 (32B-aligned) to bf16 with scale
__device__ __forceinline__ void cvt8(bf16* d, const float* p, float s) {
  const v4f* q = (const v4f*)p;
  v4f a = q[0], b = q[1];
  d[0]=to_bf16(a.x*s); d[1]=to_bf16(a.y*s); d[2]=to_bf16(a.z*s); d[3]=to_bf16(a.w*s);
  d[4]=to_bf16(b.x*s); d[5]=to_bf16(b.y*s); d[6]=to_bf16(b.z*s); d[7]=to_bf16(b.w*s);
}

// A fragment from f32 (cvt path): half 0 holds K {kb+0..15}, half 1 {kb+16..31}
__device__ __forceinline__ v16bf load_a32(const float* row, int kb, int half) {
  V16 u;
  cvt8(u.e,     row + kb + half*16,     1.0f);
  cvt8(u.e + 8, row + kb + half*16 + 8, 1.0f);
  return u.v;
}
// A fragment from pre-converted bf16 (two 16B copies, no cvt)
__device__ __forceinline__ v16bf load_afrag_bf(const bf16* row, int kb, int half) {
  V16 u;
  u.h[0] = *(const v8bf*)(row + kb + half*16);
  u.h[1] = *(const v8bf*)(row + kb + half*16 + 8);
  return u.v;
}
// B fragment from pre-converted bf16 weights: 16 contiguous K values per lane
__device__ __forceinline__ v16bf load_bfrag(const bf16* wrow, int kb, int half) {
  V16 u;
  u.h[0] = *(const v8bf*)(wrow + kb + half*16);
  u.h[1] = *(const v8bf*)(wrow + kb + half*16 + 8);
  return u.v;
}

__device__ __forceinline__ v8f wmma_bf16(v16bf a, v16bf b, v8f c) {
  return __builtin_amdgcn_wmma_f32_16x16x32_bf16(false, a, false, b, (short)0, c, false, false);
}

__device__ __forceinline__ float sel4(v4f v, int i) {
  float s = v.x;
  s = (i == 1) ? v.y : s;
  s = (i == 2) ? v.z : s;
  s = (i == 3) ? v.w : s;
  return s;
}

// ---------------------------------------------------------------------------
// Tensor Data Mover: DMA a 2-D f32 tile (tile_w x tile_h elements, row stride
// row_stride elements) from global memory into LDS at byte offset lds_off.
// D# layout per CDNA5 ISA; 6-arg builtin form.
__device__ __forceinline__ void tdm_load_2d(unsigned lds_off, const float* gaddr,
                                            unsigned tile_w, unsigned tile_h,
                                            unsigned row_stride) {
  unsigned long long ga = (unsigned long long)(uintptr_t)gaddr;
  v4u g0;
  g0.x = 0x1u;                                   // count=1, user descriptor
  g0.y = lds_off;                                // lds_addr [63:32]
  g0.z = (unsigned)(ga & 0xFFFFFFFFu);           // global_addr[95:64]
  g0.w = (unsigned)((ga >> 32) & 0x01FFFFFFu) | (2u << 30);  // addr msbs | type=2
  const unsigned td = 0x40000000u;               // huge tensor dims -> no OOB clip
  v8i g1;
  g1[0] = 0x00020000;                            // wg_mask=0, data_size=2 (4B)
  g1[1] = (int)((td & 0xFFFFu) << 16);           // [63:48]   tensor_dim0[15:0]
  g1[2] = (int)(((td >> 16) & 0xFFFFu) | ((td & 0xFFFFu) << 16));       // dim0 hi | dim1 lo
  g1[3] = (int)(((td >> 16) & 0xFFFFu) | ((tile_w & 0xFFFFu) << 16));   // dim1 hi | tile_dim0
  g1[4] = (int)(tile_h & 0xFFFFu);               // tile_dim1 | tile_dim2=0
  g1[5] = (int)row_stride;                       // tensor_dim0_stride[31:0]
  g1[6] = 0;
  g1[7] = 0;
  v4i z4 = {0, 0, 0, 0};
  v8i z8 = {0, 0, 0, 0, 0, 0, 0, 0};
  __builtin_amdgcn_tensor_load_to_lds(g0, g1, z4, z4, z8, 0);
}

// ---------------------------------------------------------------------------
// one-shot weight pre-conversion f32 -> bf16 (n multiple of 4)
__global__ void cvt_weight_kernel(const float* __restrict__ src, bf16* __restrict__ dst, int n) {
  int i = (blockIdx.x * blockDim.x + threadIdx.x) * 4;
  if (i < n) {
    v4f v = *(const v4f*)(src + i);
    dst[i]   = to_bf16(v.x);
    dst[i+1] = to_bf16(v.y);
    dst[i+2] = to_bf16(v.z);
    dst[i+3] = to_bf16(v.w);
  }
}

// per-(dst,rel) edge counts (identical for both layers -> computed once)
__global__ void count_kernel(const int* __restrict__ edst, const int* __restrict__ erel,
                             float* __restrict__ cnt) {
  int e = blockIdx.x * blockDim.x + threadIdx.x;
  if (e < N_EDGES) atomicAdd(&cnt[edst[e] * 4 + erel[e]], 1.0f);
}

// ---------------------------------------------------------------------------
// fused: msg = x[src] + ef @ We^T + be ; atomic scatter into agg[(dst*4+rel)*D + n]
// block = 16 edges; TDM DMAs the 16x64 f32 edge-feature tile into LDS, the
// block converts it to bf16 once, all 8 waves build A frags as ds b128 copies.
template<int D>
__global__ void edge_msg_scatter(const float* __restrict__ x,  const float* __restrict__ ef,
                                 const int* __restrict__ esrc, const int* __restrict__ edst,
                                 const int* __restrict__ erel,
                                 const bf16* __restrict__ We, const float* __restrict__ be,
                                 float* __restrict__ agg) {
  __shared__ float eraw[16 * 64];
  __shared__ bf16  ebf[16 * 64];
  const int wave = threadIdx.x >> 5;
  const int lane = threadIdx.x & 31;
  const int l16 = lane & 15, half = lane >> 4;
  const int ebase = blockIdx.x * 16;

  if (wave == 0) {
    tdm_load_2d((unsigned)(uintptr_t)&eraw[0], ef + (size_t)ebase * 64, 64, 16, 64);
    __builtin_amdgcn_s_wait_tensorcnt(0);
  }

  int srcs[8]; size_t segs[8];
#pragma unroll
  for (int r = 0; r < 8; ++r) {                             // rows this lane writes in C
    int e = ebase + r + 8 * half;
    srcs[r] = esrc[e];
    segs[r] = ((size_t)edst[e] * 4 + (size_t)erel[e]) * (size_t)D;
    __builtin_prefetch(&x[(size_t)srcs[r] * D], 0, 3);      // warm gather rows (L2-resident)
  }

  __syncthreads();                                          // raw tile ready
  {                                                         // block-wide cvt: each elem once
    int t = threadIdx.x * 4;
    v4f vv = *(const v4f*)&eraw[t];
    ebf[t]   = to_bf16(vv.x);
    ebf[t+1] = to_bf16(vv.y);
    ebf[t+2] = to_bf16(vv.z);
    ebf[t+3] = to_bf16(vv.w);
  }
  __syncthreads();                                          // bf16 tile ready

  const bf16* arow = &ebf[l16 * 64];
  v16bf a0 = load_afrag_bf(arow, 0,  half);                 // pure ds_load_b128
  v16bf a1 = load_afrag_bf(arow, 32, half);

  for (int nt = wave; nt < D / 16; nt += 8) {
    int n = nt * 16 + l16;
    const bf16* wrow = We + (size_t)n * 64;                 // pre-converted weights
    v16bf b0 = load_bfrag(wrow, 0,  half);
    v16bf b1 = load_bfrag(wrow, 32, half);
    v8f c = {};
    c = wmma_bf16(a0, b0, c);
    c = wmma_bf16(a1, b1, c);
    float bv = be[n];
    float xv[8];
#pragma unroll
    for (int r = 0; r < 8; ++r)                             // hoist gathers: 8 loads in flight
      xv[r] = x[(size_t)srcs[r] * D + n];
#pragma unroll
    for (int r = 0; r < 8; ++r)
      atomicAdd(&agg[segs[r] + n], c[r] + bv + xv[r]);
  }
}

// ---------------------------------------------------------------------------
// h = relu( (agg/cnt) @ Wl^T + x @ Ws^T + bl + bs )
// block = one 16-node m-tile; TDM double-buffers 16x64 f32 A-chunks into LDS,
// the block converts each chunk to bf16 once (applying the mean scale), then
// the 8 waves (covering all NOUT cols) consume it as ds b128 fragments.
template<int DIN, int NOUT>
__global__ void rgcn_combine(const float* __restrict__ agg, const float* __restrict__ cnt,
                             const float* __restrict__ x,
                             const bf16* __restrict__ Wl, const float* __restrict__ bl,
                             const bf16* __restrict__ Ws, const float* __restrict__ bs,
                             float* __restrict__ out) {
  constexpr int K4  = 4 * DIN;
  constexpr int NT  = NOUT / 128;           // n-tiles per wave (8 waves x NT x 16 = NOUT)
  constexpr int NCA = K4 / 64;              // agg chunks
  constexpr int NC  = NCA + DIN / 64;       // + self-loop x chunks
  __shared__ float raw[2][16 * 64];
  __shared__ bf16  abf[2][16 * 64];

  const int wave = threadIdx.x >> 5, lane = threadIdx.x & 31;
  const int l16 = lane & 15, half = lane >> 4;
  const int mt = blockIdx.x;
  const int m0 = mt * 16;
  const int m  = m0 + l16;

  v4f cv = *(const v4f*)(cnt + (size_t)m * 4);
  v4f invv = { 1.0f / (cv.x + 1e-10f), 1.0f / (cv.y + 1e-10f),
               1.0f / (cv.z + 1e-10f), 1.0f / (cv.w + 1e-10f) };

  int ncol[NT]; const bf16 *wlp[NT], *wsp[NT];
#pragma unroll
  for (int j = 0; j < NT; ++j) {
    ncol[j] = (wave * NT + j) * 16 + l16;
    wlp[j] = Wl + (size_t)ncol[j] * K4;
    wsp[j] = Ws + (size_t)ncol[j] * DIN;
  }

  const float* aggbase = agg + (size_t)m0 * K4;
  const float* xbase   = x   + (size_t)m0 * DIN;
  unsigned lds0 = (unsigned)(uintptr_t)&raw[0][0];
  unsigned lds1 = (unsigned)(uintptr_t)&raw[1][0];

  if (wave == 0)                                            // prime the pipeline
    tdm_load_2d(lds0, aggbase, 64, 16, K4);

  v8f c[NT] = {};
  for (int ch = 0; ch < NC; ++ch) {
    int b = ch & 1;
    if (ch + 1 < NC) {
      if (wave == 0) {                                      // issue next chunk, wait current
        int cn = ch + 1;
        const float* g = (cn < NCA) ? (aggbase + cn * 64) : (xbase + (cn - NCA) * 64);
        tdm_load_2d(b ? lds0 : lds1, g, 64, 16, (cn < NCA) ? (unsigned)K4 : (unsigned)DIN);
        __builtin_amdgcn_s_wait_tensorcnt(1);
      }
    } else {
      if (wave == 0) __builtin_amdgcn_s_wait_tensorcnt(0);
    }
    __syncthreads();                                        // raw[b] ready

    float scl = (ch < NCA) ? sel4(invv, (ch * 64) / DIN) : 1.0f;  // mean scale (uniform/chunk)
    {                                                       // block-wide cvt: each elem once
      int t = threadIdx.x * 4;
      v4f vv = *(const v4f*)&raw[b][t];
      abf[b][t]   = to_bf16(vv.x * scl);
      abf[b][t+1] = to_bf16(vv.y * scl);
      abf[b][t+2] = to_bf16(vv.z * scl);
      abf[b][t+3] = to_bf16(vv.w * scl);
    }
    __syncthreads();                                        // abf[b] ready

    const bf16* lrow = &abf[b][l16 * 64];
#pragma unroll
    for (int s = 0; s < 2; ++s) {                           // two K=32 steps per chunk
      v16bf ua = load_afrag_bf(lrow, s * 32, half);
      int kk = ch * 64 + s * 32 + half * 16;                // global k of B run
#pragma unroll
      for (int j = 0; j < NT; ++j) {
        const bf16* p = (kk < K4) ? (wlp[j] + kk) : (wsp[j] + (kk - K4));
        V16 ub;
        ub.h[0] = *(const v8bf*)p;
        ub.h[1] = *(const v8bf*)(p + 8);
        c[j] = wmma_bf16(ua, ub.v, c[j]);
      }
    }
    // no trailing barrier needed: raw[] freed at cvt (guarded by next iter's
    // barriers), abf[] double-buffered (overwritten two iterations later)
  }

#pragma unroll
  for (int j = 0; j < NT; ++j) {
    float bsum = bl[ncol[j]] + bs[ncol[j]];
#pragma unroll
    for (int r = 0; r < 8; ++r) {
      float v = fmaxf(c[j][r] + bsum, 0.0f);
      out[(size_t)(m0 + r + 8 * half) * NOUT + ncol[j]] = v;
    }
  }
}

// ---------------------------------------------------------------------------
// generic out[m, col_off+n] = [relu]( A[rows(m)] @ Wbf^T + bias ); Wbf is (N,K) bf16
// wave = one 16x64 output strip (N must be a multiple of 64)
template<int RELU>
__global__ void gemm_rows(const float* __restrict__ A, int lda,
                          const int* __restrict__ gidx, int gstride, int goff,
                          const bf16* __restrict__ W, const float* __restrict__ bias,
                          int K, int N, int M,
                          float* __restrict__ out, int ldo, int col_off) {
  const int wave = threadIdx.x >> 5, lane = threadIdx.x & 31;
  const int l16 = lane & 15, half = lane >> 4;
  int unit = blockIdx.x * 8 + wave;
  int NG = N >> 6, TM = M >> 4;
  if (unit >= TM * NG) return;
  int mt = unit / NG, ng = unit % NG;
  int m = mt * 16 + l16;
  size_t rowi = gidx ? (size_t)gidx[m * gstride + goff] : (size_t)m;
  const float* arow = A + rowi * (size_t)lda;

  int ncol[4]; const bf16* wrow[4];
#pragma unroll
  for (int j = 0; j < 4; ++j) {
    ncol[j] = ng * 64 + j * 16 + l16;
    wrow[j] = W + (size_t)ncol[j] * (size_t)K;
  }

  v8f c[4] = {};
  for (int kb = 0; kb < K; kb += 32) {
    v16bf a = load_a32(arow, kb, half);
#pragma unroll
    for (int j = 0; j < 4; ++j)
      c[j] = wmma_bf16(a, load_bfrag(wrow[j], kb, half), c[j]);
  }
#pragma unroll
  for (int j = 0; j < 4; ++j) {
    float bv = bias[ncol[j]];
#pragma unroll
    for (int r = 0; r < 8; ++r) {
      float v = c[j][r] + bv;
      if (RELU) v = fmaxf(v, 0.0f);
      out[(size_t)(mt * 16 + r + 8 * half) * ldo + col_off + ncol[j]] = v;
    }
  }
}

// dst[m, dstoff + tid] = src[idx[m], tid]   (block = 128 = row width)
__global__ void gather_rows(const float* __restrict__ src, int srcld,
                            const int* __restrict__ idx, int istride, int ioff,
                            float* __restrict__ dst, int dstld, int dstoff) {
  int m = blockIdx.x;
  int cidx = threadIdx.x;
  dst[(size_t)m * dstld + dstoff + cidx] =
      src[(size_t)idx[m * istride + ioff] * srcld + cidx];
}

// out[m] = t2[m,:] . Wf3 + bf3   (K=64, N=1)
__global__ void final_dot(const float* __restrict__ t2, const float* __restrict__ Wf3,
                          const float* __restrict__ bf3, float* __restrict__ out) {
  int m = blockIdx.x * blockDim.x + threadIdx.x;
  if (m >= BATCH) return;
  const float* r = t2 + (size_t)m * 64;
  float s = bf3[0];
#pragma unroll 8
  for (int i = 0; i < 64; ++i) s += r[i] * Wf3[i];
  out[m] = s;
}

// ---------------------------------------------------------------------------
extern "C" void kernel_launch(void* const* d_in, const int* in_sizes, int n_in,
                              void* d_out, int out_size, void* d_ws, size_t ws_size,
                              hipStream_t stream) {
  (void)in_sizes; (void)n_in; (void)out_size; (void)ws_size;
  const int*   inputs       = (const int*)d_in[0];
  const float* node_feature = (const float*)d_in[1];
  const float* edge_feature = (const float*)d_in[2];
  const int*   edge_src     = (const int*)d_in[3];
  const int*   edge_dst     = (const int*)d_in[4];
  const int*   edge_rel     = (const int*)d_in[5];
  const float *We1=(const float*)d_in[6],  *be1=(const float*)d_in[7];
  const float *Wl1=(const float*)d_in[8],  *bl1=(const float*)d_in[9];
  const float *Ws1=(const float*)d_in[10], *bs1=(const float*)d_in[11];
  const float *We2=(const float*)d_in[12], *be2=(const float*)d_in[13];
  const float *Wl2=(const float*)d_in[14], *bl2=(const float*)d_in[15];
  const float *Ws2=(const float*)d_in[16], *bs2=(const float*)d_in[17];
  const float *Wc1=(const float*)d_in[18], *bc1=(const float*)d_in[19];
  const float *Wc2=(const float*)d_in[20], *bc2=(const float*)d_in[21];
  const float *Wf1=(const float*)d_in[22], *bf1=(const float*)d_in[23];
  const float *Wf2=(const float*)d_in[24], *bf2=(const float*)d_in[25];
  const float *Wf3=(const float*)d_in[26], *bf3=(const float*)d_in[27];

  // workspace layout
  const size_t aggElems = (size_t)N_NODES * 4 * 512;        // layer-2 size (max)
  float* agg = (float*)d_ws;
  float* cnt = agg + aggElems;                              // 200000 floats
  float* h1  = cnt + (size_t)N_NODES * 4;                   // 50000 x 512
  float* h2  = h1  + (size_t)N_NODES * 512;                 // 50000 x 128
  // bf16 weight cache (converted once per launch)
  bf16* wb   = (bf16*)(h2 + (size_t)N_NODES * 128);
  bf16 *We1b = wb;              // 256*64   = 16384
  bf16 *Wl1b = We1b + 16384;    // 512*1024 = 524288
  bf16 *Ws1b = Wl1b + 524288;   // 512*256  = 131072
  bf16 *We2b = Ws1b + 131072;   // 512*64   = 32768
  bf16 *Wl2b = We2b + 32768;    // 128*2048 = 262144
  bf16 *Ws2b = Wl2b + 262144;   // 128*512  = 65536
  bf16 *Wc1b = Ws2b + 65536;    // 256*64   = 16384
  bf16 *Wc2b = Wc1b + 16384;    // 128*256  = 32768
  bf16 *Wf1b = Wc2b + 32768;    // 128*384  = 49152
  bf16 *Wf2b = Wf1b + 49152;    // 64*128   = 8192
  // head temporaries alias the (dead-after-combine2) agg region
  float* t0   = agg;                                        // 16384 x 256
  float* xcat = t0   + (size_t)BATCH * 256;                 // 16384 x 384
  float* t1   = xcat + (size_t)BATCH * 384;                 // 16384 x 128
  float* t2   = t1   + (size_t)BATCH * 128;                 // 16384 x 64

  // ---- weight pre-conversion (cheap, once per launch)
  auto cvtw = [&](const float* s, bf16* d, int n) {
    cvt_weight_kernel<<<(n / 4 + 255) / 256, 256, 0, stream>>>(s, d, n);
  };
  cvtw(We1, We1b, 256 * 64);   cvtw(Wl1, Wl1b, 512 * 1024);
  cvtw(Ws1, Ws1b, 512 * 256);  cvtw(We2, We2b, 512 * 64);
  cvtw(Wl2, Wl2b, 128 * 2048); cvtw(Ws2, Ws2b, 128 * 512);
  cvtw(Wc1, Wc1b, 256 * 64);   cvtw(Wc2, Wc2b, 128 * 256);
  cvtw(Wf1, Wf1b, 128 * 384);  cvtw(Wf2, Wf2b, 64 * 128);

  // ---- segment counts (shared by both layers)
  (void)hipMemsetAsync(cnt, 0, (size_t)N_NODES * 4 * sizeof(float), stream);
  count_kernel<<<(N_EDGES + 255) / 256, 256, 0, stream>>>(edge_dst, edge_rel, cnt);

  // ---- layer 1
  (void)hipMemsetAsync(agg, 0, (size_t)N_NODES * 4 * 256 * sizeof(float), stream);
  edge_msg_scatter<256><<<N_EDGES / 16, 256, 0, stream>>>(
      node_feature, edge_feature, edge_src, edge_dst, edge_rel, We1b, be1, agg);
  rgcn_combine<256, 512><<<N_NODES / 16, 256, 0, stream>>>(
      agg, cnt, node_feature, Wl1b, bl1, Ws1b, bs1, h1);

  // ---- layer 2
  (void)hipMemsetAsync(agg, 0, (size_t)N_NODES * 4 * 512 * sizeof(float), stream);
  edge_msg_scatter<512><<<N_EDGES / 16, 256, 0, stream>>>(
      h1, edge_feature, edge_src, edge_dst, edge_rel, We2b, be2, agg);
  rgcn_combine<512, 128><<<N_NODES / 16, 256, 0, stream>>>(
      agg, cnt, h1, Wl2b, bl2, Ws2b, bs2, h2);

  // ---- head: xcat = [h2[d1] | h2[d2] | relu(ef@Wc1^T+bc1)@Wc2^T+bc2]
  gather_rows<<<BATCH, 128, 0, stream>>>(h2, 128, inputs, 3, 0, xcat, 384, 0);
  gather_rows<<<BATCH, 128, 0, stream>>>(h2, 128, inputs, 3, 1, xcat, 384, 128);
  gemm_rows<1><<<(BATCH / 16 * (256 / 64) + 7) / 8, 256, 0, stream>>>(
      edge_feature, 64, inputs, 3, 2, Wc1b, bc1, 64, 256, BATCH, t0, 256, 0);
  gemm_rows<0><<<(BATCH / 16 * (128 / 64) + 7) / 8, 256, 0, stream>>>(
      t0, 256, nullptr, 0, 0, Wc2b, bc2, 256, 128, BATCH, xcat, 384, 256);

  // ---- MLP
  gemm_rows<1><<<(BATCH / 16 * (128 / 64) + 7) / 8, 256, 0, stream>>>(
      xcat, 384, nullptr, 0, 0, Wf1b, bf1, 384, 128, BATCH, t1, 128, 0);
  gemm_rows<1><<<(BATCH / 16 * (64 / 64) + 7) / 8, 256, 0, stream>>>(
      t1, 128, nullptr, 0, 0, Wf2b, bf2, 128, 64, BATCH, t2, 64, 0);
  final_dot<<<(BATCH + 255) / 256, 256, 0, stream>>>(t2, Wf3, bf3, (float*)d_out);
}
